// ImageScaler_85822036509001
// MI455X (gfx1250) — matrix operand order repaired and checked by reading, not compiled, not run on hardware
//
#include <hip/hip_runtime.h>
#include <cstddef>

// ---------------- problem constants ----------------
#define NREF    200000
#define NIMG_   500
#define DMETA_  24
#define WIDTH_  32
#define HID_    64
#define DEPTH_  20
#define NTILES  (NREF / 16)        // 12500, exact; pairs = 6250

// ---------------- LDS layout (bytes) ----------------
#define OFF_W    0                         // 20*8192 = 163840
#define OFF_B1   163840                    // 20*64 f32 = 5120
#define OFF_B2   168960                    // 20*32 f32 = 2560
#define OFF_WIN  171520                    // 2*1024 bf16 input-layer A blocks
#define OFF_BIN  173568                    // 32 f32 input bias
#define LDS_BYTES 173696

typedef __bf16 v16bf __attribute__((ext_vector_type(16)));
typedef float  v8f   __attribute__((ext_vector_type(8)));

static __device__ __forceinline__ v8f wmma_bf16(v16bf a, v16bf b, v8f c) {
    // D = A(16x32 bf16) x B(32x16 bf16) + C(16x16 f32)
    return __builtin_amdgcn_wmma_f32_16x16x32_bf16(false, a, false, b,
                                                   (short)0, c, false, false);
}

// Guaranteed single-instruction pack of two f32 into one dword of bf16 pairs.
static __device__ __forceinline__ unsigned pk_bf16(float lo, float hi) {
    unsigned r;
    asm("v_cvt_pk_bf16_f32 %0, %1, %2" : "=v"(r) : "v"(lo), "v"(hi));
    return r;
}

// VALU xor-16 lane exchange (v_permlanex16_b32): identity selects swap the two
// 16-lane halves of the wave32 — avoids ds_bpermute / DScnt entirely.
static __device__ __forceinline__ unsigned xor16u(unsigned v) {
#if __has_builtin(__builtin_amdgcn_permlanex16)
    return (unsigned)__builtin_amdgcn_permlanex16((int)v, (int)v,
                                                  0x76543210, (int)0xfedcba98,
                                                  true, false);
#else
    return (unsigned)__shfl_xor((int)v, 16, 32);
#endif
}
static __device__ __forceinline__ float xor16f(float v) {
    union { float f; unsigned u; } a; a.f = v; a.u = xor16u(a.u); return a.f;
}

static __device__ __forceinline__ unsigned hashu(unsigned x) {
    x ^= x >> 16; x *= 0x7feb352dU;
    x ^= x >> 15; x *= 0x846ca68bU;
    x ^= x >> 16;
    return x;
}

// Build a bf16 B operand (32x16, K=feature, N=batch) from two f32 C/D tiles.
// 8x v_cvt_pk_bf16_f32 + 8x v_permlanex16 + 8x v_cndmask — pure VALU,
// co-executes with the XDL WMMA pipe.
static __device__ __forceinline__ v16bf make_B(const v8f t0, const v8f t1, int half) {
    unsigned p0[4], p1[4];
#pragma unroll
    for (int j = 0; j < 4; ++j) {
        p0[j] = pk_bf16(t0[2 * j], t0[2 * j + 1]);
        p1[j] = pk_bf16(t1[2 * j], t1[2 * j + 1]);
    }
    union { unsigned u[8]; v16bf v; } B;
#pragma unroll
    for (int j = 0; j < 4; ++j) {
        unsigned q0 = xor16u(p0[j]);        // remote t0 pack
        unsigned q1 = xor16u(p1[j]);        // remote t1 pack
        B.u[j]     = (half == 0) ? p0[j] : q1;       // K = half*16 + [0..7]
        B.u[4 + j] = (half == 0) ? q0    : p1[j];    // K = half*16 + [8..15]
    }
    return B.v;
}

// One-time cooperative swizzle of all weights into per-lane A-operand order.
// A-tile element mapping: lane -> m = lane&15, half = lane>>4;
// elem i -> k = half*16 + i.
static __device__ void stage_weights(char* s,
                                     const float* w1, const float* b1,
                                     const float* w2, const float* b2,
                                     const float* win, const float* bin,
                                     int din, int tid, int nthr)
{
    __bf16* wl = (__bf16*)(s + OFF_W);
    for (int e = tid; e < DEPTH_ * 8 * 512; e += nthr) {
        int L   = e >> 12;
        int rem = e & 4095;
        int blk = rem >> 9;
        int pos = rem & 511;
        int lane = pos >> 4, i = pos & 15;
        int m = lane & 15, half = lane >> 4;
        int k = half * 16 + i;
        float v;
        if (blk < 4) {                 // GEMM1: A[m][k] = W1[k][blk*16+m]   (W1: 32x64)
            v = w1[L * (WIDTH_ * HID_) + k * HID_ + blk * 16 + m];
        } else {                       // GEMM2 tile t, k-slice kk           (W2: 64x32)
            int t = (blk - 4) >> 1, kk = (blk - 4) & 1;
            v = w2[L * (HID_ * WIDTH_) + (kk * 32 + k) * WIDTH_ + t * 16 + m];
        }
        wl[e] = (__bf16)v;
    }
    __bf16* wi = (__bf16*)(s + OFF_WIN);
    for (int e = tid; e < 2 * 512; e += nthr) {
        int blk = e >> 9, pos = e & 511;
        int lane = pos >> 4, i = pos & 15;
        int m = lane & 15, half = lane >> 4;
        int k = half * 16 + i;
        wi[e] = (__bf16)((k < din) ? win[k * WIDTH_ + blk * 16 + m] : 0.0f);
    }
    float* f1 = (float*)(s + OFF_B1);
    for (int e = tid; e < DEPTH_ * HID_; e += nthr)   f1[e] = b1[e];
    float* f2 = (float*)(s + OFF_B2);
    for (int e = tid; e < DEPTH_ * WIDTH_; e += nthr) f2[e] = b2[e];
    float* fb = (float*)(s + OFF_BIN);
    for (int e = tid; e < WIDTH_; e += nthr)          fb[e] = bin[e];
}

// 20 residual blocks for TWO row-tiles at once: weights (A operand) and biases
// are loaded from LDS once and feed both tiles' WMMAs; the two independent
// accumulator chains interleave to hide WMMA->pack->WMMA latency.
static __device__ __forceinline__ void run_mlp2(const char* s, int lane,
                                                v8f& hA0, v8f& hA1, v8f& hB0, v8f& hB1)
{
    const int half = lane >> 4, mbase = half * 8;
    for (int L = 0; L < DEPTH_; ++L) {
        const char*  wb = s + OFF_W + L * 8192;
        const float* p1 = (const float*)(s + OFF_B1) + L * HID_;
        const float* p2 = (const float*)(s + OFF_B2) + L * WIDTH_;

        v16bf bhA = make_B(hA0, hA1, half);
        v16bf bhB = make_B(hB0, hB1, half);
        v8f uA[4], uB[4];
#pragma unroll
        for (int t = 0; t < 4; ++t) {
            v8f  c = *(const v8f*)(p1 + t * 16 + mbase);
            v16bf a = *(const v16bf*)(wb + t * 1024 + (size_t)lane * 32);
            uA[t] = wmma_bf16(a, bhA, c);
            uB[t] = wmma_bf16(a, bhB, c);
        }
#pragma unroll
        for (int t = 0; t < 4; ++t)
#pragma unroll
            for (int r = 0; r < 8; ++r) {
                uA[t][r] = fmaxf(uA[t][r], 0.0f);
                uB[t][r] = fmaxf(uB[t][r], 0.0f);
            }

        v16bf b0A = make_B(uA[0], uA[1], half);
        v16bf b1A = make_B(uA[2], uA[3], half);
        v16bf b0B = make_B(uB[0], uB[1], half);
        v16bf b1B = make_B(uB[2], uB[3], half);

        {
            v16bf a4 = *(const v16bf*)(wb + 4 * 1024 + (size_t)lane * 32);
            v16bf a5 = *(const v16bf*)(wb + 5 * 1024 + (size_t)lane * 32);
            v8f  bb = *(const v8f*)(p2 + mbase);
            v8f cA = hA0 + bb, cB = hB0 + bb;
            cA = wmma_bf16(a4, b0A, cA);  cB = wmma_bf16(a4, b0B, cB);
            cA = wmma_bf16(a5, b1A, cA);  cB = wmma_bf16(a5, b1B, cB);
            hA0 = cA; hB0 = cB;
        }
        {
            v16bf a6 = *(const v16bf*)(wb + 6 * 1024 + (size_t)lane * 32);
            v16bf a7 = *(const v16bf*)(wb + 7 * 1024 + (size_t)lane * 32);
            v8f  bb = *(const v8f*)(p2 + 16 + mbase);
            v8f cA = hA1 + bb, cB = hB1 + bb;
            cA = wmma_bf16(a6, b0A, cA);  cB = wmma_bf16(a6, b0B, cB);
            cA = wmma_bf16(a7, b1A, cA);  cB = wmma_bf16(a7, b1B, cB);
            hA1 = cA; hB1 = cB;
        }
    }
}

// Per-lane input B operand: K = half*16 + i over padded input vector
// (metadata 24 cols [+ iobs + sigiobs for pass1], zero-padded to 32).
static __device__ __forceinline__ v16bf load_input_B(const float* meta, const float* iobs,
                                                     const float* sig, int row, int half,
                                                     bool with_obs)
{
    const float* mrow = meta + (size_t)row * DMETA_;
    float f[16];
    if (half == 0) {
        v8f a = *(const v8f*)(mrow);
        v8f b = *(const v8f*)(mrow + 8);
#pragma unroll
        for (int r = 0; r < 8; ++r) { f[r] = a[r]; f[8 + r] = b[r]; }
    } else {
        v8f a = *(const v8f*)(mrow + 16);
#pragma unroll
        for (int r = 0; r < 8; ++r) f[r] = a[r];
        f[8] = with_obs ? iobs[row] : 0.0f;
        f[9] = with_obs ? sig[row]  : 0.0f;
#pragma unroll
        for (int r = 10; r < 16; ++r) f[r] = 0.0f;
    }
    union { unsigned u[8]; v16bf v; } B;
#pragma unroll
    for (int j = 0; j < 8; ++j) B.u[j] = pk_bf16(f[2 * j], f[2 * j + 1]);
    return B.v;
}

template <int PASS>
__global__ __launch_bounds__(256)
void mlp_pass(const float* __restrict__ meta, const float* __restrict__ iobs,
              const float* __restrict__ sig,
              const float* __restrict__ win, const float* __restrict__ binp,
              const float* __restrict__ w1, const float* __restrict__ b1,
              const float* __restrict__ w2, const float* __restrict__ b2,
              const float* __restrict__ wout, const float* __restrict__ bout,
              const int* __restrict__ ids, const float* __restrict__ pooled,
              float* __restrict__ hbuf, float* __restrict__ outz,
              float* __restrict__ outkl)
{
    extern __shared__ __align__(32) char smem[];
    const int tid = threadIdx.x;
    stage_weights(smem, w1, b1, w2, b2, win, binp,
                  (PASS == 1) ? (DMETA_ + 2) : DMETA_, tid, blockDim.x);
    __syncthreads();

    const int lane = tid & 31, warp = tid >> 5;
    const int half = lane >> 4, mbase = half * 8, ncol = lane & 15;
    const int wid = blockIdx.x * (blockDim.x >> 5) + warp;
    const int nw  = gridDim.x * (blockDim.x >> 5);

    // tile-invariant hoists
    const float* binL = (const float*)(smem + OFF_BIN);
    const v8f bin0 = *(const v8f*)(binL + mbase);
    const v8f bin1 = *(const v8f*)(binL + 16 + mbase);

    float wl0[8], wl1[8], wh0[8], wh1[8], bo0 = 0.f, bo1 = 0.f;
    if (PASS == 2) {
#pragma unroll
        for (int r = 0; r < 8; ++r) {
            wl0[r] = wout[(mbase + r) * 2 + 0];
            wh0[r] = wout[(mbase + r) * 2 + 1];
            wl1[r] = wout[(16 + mbase + r) * 2 + 0];
            wh1[r] = wout[(16 + mbase + r) * 2 + 1];
        }
        bo0 = bout[0]; bo1 = bout[1];
    }

    for (int pair = wid; pair < NTILES / 2; pair += nw) {
        const int rowA = (pair * 2) * 16 + ncol;
        const int rowB = (pair * 2 + 1) * 16 + ncol;

        v16bf bIA = load_input_B(meta, iobs, sig, rowA, half, PASS == 1);
        v16bf bIB = load_input_B(meta, iobs, sig, rowB, half, PASS == 1);

        v8f hA0 = bin0, hA1 = bin1, hB0 = bin0, hB1 = bin1;
        if (PASS == 2) {                             // + pooled[image_id[row]]
            const float* pa = pooled + (size_t)ids[rowA] * WIDTH_;
            const float* pb = pooled + (size_t)ids[rowB] * WIDTH_;
            hA0 += *(const v8f*)(pa + mbase);
            hA1 += *(const v8f*)(pa + 16 + mbase);
            hB0 += *(const v8f*)(pb + mbase);
            hB1 += *(const v8f*)(pb + 16 + mbase);
        }
        {
            v16bf aw0 = *(const v16bf*)(smem + OFF_WIN + (size_t)lane * 32);
            v16bf aw1 = *(const v16bf*)(smem + OFF_WIN + 1024 + (size_t)lane * 32);
            hA0 = wmma_bf16(aw0, bIA, hA0);  hB0 = wmma_bf16(aw0, bIB, hB0);
            hA1 = wmma_bf16(aw1, bIA, hA1);  hB1 = wmma_bf16(aw1, bIB, hB1);
        }

        run_mlp2(smem, lane, hA0, hA1, hB0, hB1);

        if (PASS == 1) {
            float* ha = hbuf + (size_t)rowA * 32;
            float* hb = hbuf + (size_t)rowB * 32;
            *(v8f*)(ha + mbase)      = hA0;
            *(v8f*)(ha + 16 + mbase) = hA1;
            *(v8f*)(hb + mbase)      = hB0;
            *(v8f*)(hb + 16 + mbase) = hB1;
        } else {
#pragma unroll
            for (int j2 = 0; j2 < 2; ++j2) {
                const int row = j2 ? rowB : rowA;
                const v8f s0 = j2 ? hB0 : hA0;
                const v8f s1 = j2 ? hB1 : hA1;
                // params = s @ w_out + b_out (dot over 32 feats, split across lane pair)
                float pl = 0.f, ps = 0.f;
#pragma unroll
                for (int r = 0; r < 8; ++r) {
                    pl += s0[r] * wl0[r] + s1[r] * wl1[r];
                    ps += s0[r] * wh0[r] + s1[r] * wh1[r];
                }
                float loc = pl + xor16f(pl) + bo0;
                float sp  = ps + xor16f(ps) + bo1;
                float scale = ((sp > 20.f) ? sp : log1pf(__expf(sp))) + 1e-12f;
                float logsc = __logf(scale);

                float klacc = 0.f;
                const int i0 = half * 16;            // 16 MC samples per lane of the pair
                for (int j = 0; j < 16; ++j) {
                    unsigned idx = (unsigned)row * 32u + (unsigned)(i0 + j);
                    unsigned r1 = hashu(idx ^ 0x9E3779B9u);
                    unsigned r2 = hashu(idx + 0x7F4A7C15u);
                    float u1 = ((float)r1 + 0.5f) * 2.3283064e-10f;
                    float u2 = ((float)r2 + 0.5f) * 2.3283064e-10f;
                    float nz = sqrtf(-2.0f * __logf(u1)) * __cosf(6.28318530718f * u2);
                    float zz = loc + scale * nz;
                    outz[(size_t)row * 32 + i0 + j] = zz;
                    // log_q - log_p = -0.5 n^2 - log(scale) - 0.91894 + |z| + 0.69315
                    klacc += -0.5f * nz * nz + fabsf(zz) - logsc - 0.22579135f;
                }
                float tot = klacc + xor16f(klacc);
                if (half == 0) outkl[row] = 0.01f * tot * (1.0f / 32.0f);
            }
        }
    }
}

// Deterministic segment mean: ids are sorted, so image i owns rows
// [sum(counts[:i]), +counts[i]). Fixed-order strided sums + fixed tree reduce.
__global__ __launch_bounds__(256)
void pool_kernel(const float* __restrict__ hbuf, const int* __restrict__ counts,
                 float* __restrict__ pooled)
{
    __shared__ float red[8][32];
    __shared__ int s_start;
    const int img = blockIdx.x;
    const int feat = threadIdx.x;  // 0..31
    const int sub  = threadIdx.y;  // 0..7
    if (feat == 0 && sub == 0) {
        int st = 0;
        for (int j = 0; j < img; ++j) st += counts[j];
        s_start = st;
    }
    __syncthreads();
    const int start = s_start;
    const int cnt = counts[img];
    float sum = 0.f;
    for (int r = sub; r < cnt; r += 8)
        sum += hbuf[(size_t)(start + r) * 32 + feat];
    red[sub][feat] = sum;
    __syncthreads();
    for (int s = 4; s > 0; s >>= 1) {
        if (sub < s) red[sub][feat] += red[sub + s][feat];
        __syncthreads();
    }
    if (sub == 0) pooled[img * 32 + feat] = red[0][feat] / (float)((cnt > 1) ? cnt : 1);
}

extern "C" void kernel_launch(void* const* d_in, const int* in_sizes, int n_in,
                              void* d_out, int out_size, void* d_ws, size_t ws_size,
                              hipStream_t stream)
{
    const float* meta  = (const float*)d_in[0];
    const float* iobs  = (const float*)d_in[1];
    const float* sig   = (const float*)d_in[2];
    const float* w_img = (const float*)d_in[3];
    const float* b_img = (const float*)d_in[4];
    const float* w_sc  = (const float*)d_in[5];
    const float* b_sc  = (const float*)d_in[6];
    const float* w1    = (const float*)d_in[7];
    const float* b1    = (const float*)d_in[8];
    const float* w2    = (const float*)d_in[9];
    const float* b2    = (const float*)d_in[10];
    const float* wout  = (const float*)d_in[11];
    const float* bout  = (const float*)d_in[12];
    const int*   ids   = (const int*)d_in[13];
    const int*   cnts  = (const int*)d_in[14];
    (void)in_sizes; (void)n_in; (void)out_size; (void)ws_size;

    char*  ws     = (char*)d_ws;
    float* hbuf   = (float*)ws;                                   // N*32 f32 = 25.6 MB
    float* pooled = (float*)(ws + (size_t)NREF * 32 * sizeof(float));

    float* outz  = (float*)d_out;                                 // (N,32) z.T
    float* outkl = outz + (size_t)NREF * 32;                      // (N,) kl

    (void)hipFuncSetAttribute(reinterpret_cast<const void*>(&mlp_pass<1>),
                              hipFuncAttributeMaxDynamicSharedMemorySize, LDS_BYTES);
    (void)hipFuncSetAttribute(reinterpret_cast<const void*>(&mlp_pass<2>),
                              hipFuncAttributeMaxDynamicSharedMemorySize, LDS_BYTES);

    mlp_pass<1><<<200, 256, LDS_BYTES, stream>>>(meta, iobs, sig, w_img, b_img,
                                                 w1, b1, w2, b2,
                                                 nullptr, nullptr, nullptr, nullptr,
                                                 hbuf, nullptr, nullptr);
    pool_kernel<<<NIMG_, dim3(32, 8), 0, stream>>>(hbuf, cnts, pooled);
    mlp_pass<2><<<200, 256, LDS_BYTES, stream>>>(meta, iobs, sig, w_sc, b_sc,
                                                 w1, b1, w2, b2,
                                                 wout, bout, ids, pooled,
                                                 nullptr, outz, outkl);
}
